// SimpleSelfAttention_11819749998849
// MI455X (gfx1250) — matrix operand order repaired and checked by reading, not compiled, not run on hardware
//
#include <hip/hip_runtime.h>
#include <hip/hip_bf16.h>

typedef __attribute__((ext_vector_type(16))) _Float16 v16h;
typedef __attribute__((ext_vector_type(8)))  _Float16 v8h;
typedef __attribute__((ext_vector_type(8)))  float    v8f;

#define BB 2
#define HH 16
#define SS 2048
#define DD 64

// ---------------------------------------------------------------------------
// Kernel 1: fused q/k projection (f32 VALU) + interleaved RoPE, emit f16 q/k.
// One thread per (position, rotation-pair i). Computes channels 2i and 2i+1
// of both q and k (64 MACs each, 256 total), rotates, stores f16.
// The 1/sqrt(D) = 0.125 score scale is folded into q here, so the WMMA
// kernel can store raw accumulators (no post-scale VALU, no extra hazards).
// ---------------------------------------------------------------------------
__global__ void proj_rope_f16_kernel(const float* __restrict__ x,
                                     const int*   __restrict__ pos,
                                     const float* __restrict__ Wq,
                                     const float* __restrict__ Wk,
                                     _Float16* __restrict__ qh,
                                     _Float16* __restrict__ kh) {
    const int pairs = DD / 2;                       // 32
    int tid = blockIdx.x * blockDim.x + threadIdx.x;
    int p = tid / pairs;                            // flat position 0..B*H*S-1
    int i = tid - p * pairs;                        // pair index 0..31

    const float* __restrict__ xr  = x  + (size_t)p * DD;
    const float* __restrict__ wq0 = Wq + (size_t)(2 * i)     * DD;
    const float* __restrict__ wq1 = Wq + (size_t)(2 * i + 1) * DD;
    const float* __restrict__ wk0 = Wk + (size_t)(2 * i)     * DD;
    const float* __restrict__ wk1 = Wk + (size_t)(2 * i + 1) * DD;

    float q0 = 0.f, q1 = 0.f, k0 = 0.f, k1 = 0.f;
#pragma unroll
    for (int d = 0; d < DD; ++d) {
        float xv = xr[d];
        q0 = fmaf(xv, wq0[d], q0);
        q1 = fmaf(xv, wq1[d], q1);
        k0 = fmaf(xv, wk0[d], k0);
        k1 = fmaf(xv, wk1[d], k1);
    }

    // inv_freq = 10000^(-2i/D) = exp(-2i * ln(10000)/D)
    float inv_freq = expf(-(float)(2 * i) * (9.210340371976184f / (float)DD));
    float ang = (float)pos[p] * inv_freq;
    float s = sinf(ang);
    float c = cosf(ang);

    const float scale = 0.125f;                     // 1/sqrt(64), folded into q
    size_t o = (size_t)p * DD + 2 * i;
    qh[o]     = (_Float16)((q0 * c - q1 * s) * scale);
    qh[o + 1] = (_Float16)((q1 * c + q0 * s) * scale);
    kh[o]     = (_Float16)(k0 * c - k1 * s);
    kh[o + 1] = (_Float16)(k1 * c + k0 * s);
}

// ---------------------------------------------------------------------------
// Fragment loaders for V_WMMA_F32_16X16X32_F16 (wave32).
//
// A-matrix 16x32 (MxK), per ISA 7.12.2:
//   lane L: M = L&15 ; element j: K = ((L>>4)*8) + (j<8 ? j : j+8)
//   -> two 16-byte chunks at K = kb and K = kb+16, kb = (L>>4)*8 + koff.
// ---------------------------------------------------------------------------
__device__ __forceinline__ v16h load_fragA(const _Float16* __restrict__ tile,
                                           int lane, int koff) {
    int row = lane & 15;
    int kb  = ((lane >> 4) << 3) + koff;
    const _Float16* p = tile + row * DD + kb;
    v8h lo = *reinterpret_cast<const v8h*>(p);        // K = kb .. kb+7
    v8h hi = *reinterpret_cast<const v8h*>(p + 16);   // K = kb+16 .. kb+23
    v16h r;
#pragma unroll
    for (int j = 0; j < 8; ++j) { r[j] = lo[j]; r[j + 8] = hi[j]; }
    return r;
}

// B-matrix 32x16 (KxN): lane L: N = L&15 ; element j: K = ((L>>4)*8) + (j<8 ? j : j+8)
//   -> two 16-byte chunks at K = kc and K = kc+16. Since B[k][n] = k_mat[n][k], we read
//   row (colTile + N) of row-major f16 k directly.
__device__ __forceinline__ v16h load_fragB(const _Float16* __restrict__ tile,
                                           int lane, int koff) {
    int trow = lane & 15;
    int kc   = ((lane >> 4) << 3) + koff;
    const _Float16* p = tile + trow * DD + kc;
    v8h lo = *reinterpret_cast<const v8h*>(p);        // K = kc .. kc+7
    v8h hi = *reinterpret_cast<const v8h*>(p + 16);   // K = kc+16 .. kc+23
    v16h r;
#pragma unroll
    for (int j = 0; j < 8; ++j) { r[j] = lo[j]; r[j + 8] = hi[j]; }
    return r;
}

// ---------------------------------------------------------------------------
// Kernel 2: scores = (q_scaled @ k^T) per (b,h).   (scale pre-folded into q)
// Block = 128 threads (4 waves). Block tile 64x64; each wave a 32x32 tile
// as 2x2 WMMA accumulators; K=64 -> 2 chained WMMAs per accumulator.
// ---------------------------------------------------------------------------
__global__ void scores_wmma_kernel(const _Float16* __restrict__ qh,
                                   const _Float16* __restrict__ kh,
                                   float* __restrict__ out) {
    int lane = threadIdx.x & 31;
    int w    = threadIdx.x >> 5;        // wave 0..3
    int wm   = w >> 1;                  // wave row in block tile
    int wn   = w & 1;                   // wave col in block tile
    int bz   = blockIdx.z;              // flat (b*H + h)

    const _Float16* __restrict__ qb = qh + (size_t)bz * SS * DD;
    const _Float16* __restrict__ kb = kh + (size_t)bz * SS * DD;

    int rowT = blockIdx.y * 64 + wm * 32;   // q rows   (s)
    int colT = blockIdx.x * 64 + wn * 32;   // k rows   (t)

    v16h a[2][2], b[2][2];
#pragma unroll
    for (int m2 = 0; m2 < 2; ++m2)
#pragma unroll
        for (int kk = 0; kk < 2; ++kk)
            a[m2][kk] = load_fragA(qb + (size_t)(rowT + m2 * 16) * DD, lane, kk * 32);
#pragma unroll
    for (int n2 = 0; n2 < 2; ++n2)
#pragma unroll
        for (int kk = 0; kk < 2; ++kk)
            b[n2][kk] = load_fragB(kb + (size_t)(colT + n2 * 16) * DD, lane, kk * 32);

    v8f c[2][2];
#pragma unroll
    for (int m2 = 0; m2 < 2; ++m2)
#pragma unroll
        for (int n2 = 0; n2 < 2; ++n2) {
            v8f acc = {};
            acc = __builtin_amdgcn_wmma_f32_16x16x32_f16(
                false, a[m2][0], false, b[n2][0], (short)0, acc, false, false);
            acc = __builtin_amdgcn_wmma_f32_16x16x32_f16(
                false, a[m2][1], false, b[n2][1], (short)0, acc, false, false);
            c[m2][n2] = acc;
        }

    size_t obase = (size_t)bz * SS * SS;
#pragma unroll
    for (int m2 = 0; m2 < 2; ++m2) {
        int rbase = rowT + m2 * 16 + ((lane >> 4) << 3);
#pragma unroll
        for (int n2 = 0; n2 < 2; ++n2) {
            int col = colT + n2 * 16 + (lane & 15);
#pragma unroll
            for (int r = 0; r < 8; ++r)
                out[obase + (size_t)(rbase + r) * SS + col] = c[m2][n2][r];
        }
    }
}

extern "C" void kernel_launch(void* const* d_in, const int* in_sizes, int n_in,
                              void* d_out, int out_size, void* d_ws, size_t ws_size,
                              hipStream_t stream) {
    (void)in_sizes; (void)n_in; (void)out_size; (void)ws_size;

    const float* x   = (const float*)d_in[0];
    const int*   pos = (const int*)  d_in[1];
    const float* Wq  = (const float*)d_in[2];
    const float* Wk  = (const float*)d_in[3];
    float* out = (float*)d_out;

    const int P = BB * HH * SS;                 // 65536 positions
    _Float16* qh = (_Float16*)d_ws;             // [P, 64] f16 (scaled by 0.125)
    _Float16* kh = qh + (size_t)P * DD;         // [P, 64] f16  (16.8 MB total)

    // Phase 1: projection + RoPE -> f16 q/k
    {
        int threads = P * (DD / 2);             // 2,097,152
        dim3 block(256);
        dim3 grid(threads / 256);
        proj_rope_f16_kernel<<<grid, block, 0, stream>>>(x, pos, Wq, Wk, qh, kh);
    }

    // Phase 2: WMMA scores
    {
        dim3 block(128);                        // 4 waves
        dim3 grid(SS / 64, SS / 64, BB * HH);   // (32, 32, 32)
        scores_wmma_kernel<<<grid, block, 0, stream>>>(qh, kh, out);
    }
}
